// _GRUDHead_32581621908146
// MI455X (gfx1250) — matrix operand order repaired and checked by reading, not compiled, not run on hardware
//
#include <hip/hip_runtime.h>
#include <hip/hip_bf16.h>
#include <stdint.h>

// ---------------------------------------------------------------------------
// GRU-D (2 layer) + head for MI455X (gfx1250), bf16 WMMA compute, f32 accum.
//
// B=256, T=512, D=64, E=32 (DIN0=96), H=256, HOR=24.
// Strategy:
//   * time-parallel prep kernels: x_filled (bf16), mean-delta, weight packing
//     into CDNA5 WMMA B-fragment layout (ISA 7.12.2).
//   * persistent scan kernel: 16 WGs (one 16-row batch tile each), 8 wave32s
//     per WG, each wave owns two 16-col N tiles of H=256. All per-step GEMMs
//     run as v_wmma_f32_16x16x32_bf16; activations live in LDS; weights are
//     streamed from L2 every step (packed weights ~1.5MB << 192MB L2).
//   * an opaque zero OFFSET (not the pointer) is laundered through inline asm
//     each time step: LICM cannot hoist the loop-invariant B-fragment loads
//     (rev1 failure: 150+ hoisted fragments spilled to scratch), while the
//     pointers stay kernarg-derived so the compiler emits global_load_b128
//     with SGPR base instead of flat_load_b128 (rev2 regression).
// ---------------------------------------------------------------------------

typedef __attribute__((ext_vector_type(16))) __bf16 v16bf;
typedef __attribute__((ext_vector_type(8)))  __bf16 v8bf;
typedef __attribute__((ext_vector_type(8)))  float  v8f;

#define WMMA_BF16(a, b, c) \
  __builtin_amdgcn_wmma_f32_16x16x32_bf16(false, (a), false, (b), (short)0, (c), false, false)

__device__ __forceinline__ v8f zero8() {
  v8f z = {0.f, 0.f, 0.f, 0.f, 0.f, 0.f, 0.f, 0.f};
  return z;
}

__device__ __forceinline__ float sigmoidf_(float x) {
  return 1.0f / (1.0f + __expf(-x));
}

// Opaque zero, re-materialized every call site (volatile): adding it to the
// weight indices defeats LICM without destroying global-addrspace inference.
__device__ __forceinline__ size_t opaque_zero() {
  size_t z = 0;
  asm volatile("" : "+s"(z));
  return z;
}

// Build an A fragment (16x32 bf16, ISA 7.12.2 A layout) from a row-major
// [16][pitch] bf16 buffer. lane<16: K {0..7},{16..23}; lane>=16: K {8..15},{24..31}.
__device__ __forceinline__ v16bf make_a(const __bf16* buf, int pitch, int kcol) {
  const int lane = threadIdx.x & 31;
  const int m    = lane & 15;
  const int sel  = (lane >> 4) << 3;           // 0 or 8
  const __bf16* p = buf + m * pitch + kcol + sel;
  v8bf lo = *reinterpret_cast<const v8bf*>(p);
  v8bf hi = *reinterpret_cast<const v8bf*>(p + 16);
  return __builtin_shufflevector(lo, hi, 0, 1, 2, 3, 4, 5, 6, 7,
                                 8, 9, 10, 11, 12, 13, 14, 15);
}

// Load a B fragment from a packed weight tile (512 bf16 per 32x16 tile;
// lane-contiguous 32B per lane).
__device__ __forceinline__ v16bf load_b(const __bf16* tile) {
  const int lane = threadIdx.x & 31;
  return *reinterpret_cast<const v16bf*>(tile + lane * 16);
}

// ---------------------------------------------------------------------------
// Pack W [K,N] (row-major f32, fan_in x fan_out) into bf16 B-fragment tiles.
// Tile (kt,nt) holds a 32x16 block; within a tile, word i = lane*8 + v holds
// bf16 pair (k = kt*32 + (lane>>4)*8 + (v>>2)*8 + 2v + {0,1}, n = nt*16 + (lane&15)).
// ---------------------------------------------------------------------------
__global__ void pack_b_kernel(const float* __restrict__ W, int K, int N,
                              __bf16* __restrict__ dst) {
  const int words = (K / 32) * (N / 16) * 256;
  int i = blockIdx.x * blockDim.x + threadIdx.x;
  if (i >= words) return;
  const int v    = i & 7;
  const int lane = (i >> 3) & 31;
  const int tile = i >> 8;
  const int NT   = N / 16;
  const int kt   = tile / NT;
  const int nt   = tile % NT;
  const int n    = nt * 16 + (lane & 15);
  const int k    = kt * 32 + ((lane >> 4) << 3) + ((v >> 2) << 3) + v * 2;
  dst[i * 2 + 0] = (__bf16)W[(k + 0) * N + n];
  dst[i * 2 + 1] = (__bf16)W[(k + 1) * N + n];
}

// ---------------------------------------------------------------------------
// x_filled precompute (time-parallel): xf[t][btile][m][c], c in [0,96), bf16.
// c<64: mask ? x : exp(-relu(delta*gx0))*x_last ; c>=64: station embedding
// (emb cols have mask=1, so filled value is the embedding itself).
// ---------------------------------------------------------------------------
__global__ void xfill_kernel(const float* __restrict__ x,
                             const float* __restrict__ mask,
                             const float* __restrict__ delta,
                             const float* __restrict__ xlast,
                             const int*   __restrict__ sidx,
                             const float* __restrict__ emb,
                             const float* __restrict__ gx0,
                             __bf16* __restrict__ xf) {
  const long total = 512L * 256L * 96L;
  long i = (long)blockIdx.x * blockDim.x + threadIdx.x;
  if (i >= total) return;
  const int  c  = (int)(i % 96);
  const long r  = i / 96;
  const int  m  = (int)(r & 15);
  const int  bt = (int)((r >> 4) & 15);
  const int  t  = (int)(r >> 8);
  const int  b  = bt * 16 + m;
  float val;
  if (c < 64) {
    const long s = ((long)b * 512 + t) * 64 + c;
    const float dec = __expf(-fmaxf(delta[s] * gx0[c], 0.f)) * xlast[s];
    val = (mask[s] > 0.f) ? x[s] : dec;
  } else {
    val = emb[sidx[b] * 32 + (c - 64)];
  }
  xf[i] = (__bf16)val;
}

// mean(delta_c) over DIN0=96 cols (E cols are zero) -> md0[b*T + t]
__global__ void md_kernel(const float* __restrict__ delta,
                          float* __restrict__ md0) {
  int i = blockIdx.x * blockDim.x + threadIdx.x;
  if (i >= 256 * 512) return;
  const int b = i / 512, t = i % 512;
  const float* d = delta + ((long)b * 512 + t) * 64;
  float s = 0.f;
#pragma unroll 8
  for (int c = 0; c < 64; ++c) s += d[c];
  md0[i] = s * (1.0f / 96.0f);
}

// ---------------------------------------------------------------------------
// Persistent scan: 16 blocks x 256 threads. Block bt owns batch rows
// [16*bt, 16*bt+16). Wave w owns N tiles {2w, 2w+1} of H=256.
// ---------------------------------------------------------------------------
__global__ __launch_bounds__(256, 1) void grud_scan_kernel(
    const __bf16* __restrict__ xf,  const float* __restrict__ md0,
    const float* __restrict__ gh0,
    const __bf16* __restrict__ wr0, const __bf16* __restrict__ wz0,
    const __bf16* __restrict__ wh0,
    const float* __restrict__ br0,  const float* __restrict__ bz0,
    const float* __restrict__ bh0,
    const __bf16* __restrict__ wr1, const __bf16* __restrict__ wz1,
    const __bf16* __restrict__ wh1,
    const float* __restrict__ br1,  const float* __restrict__ bz1,
    const float* __restrict__ bh1,
    const __bf16* __restrict__ wp1, const float* __restrict__ bp1,
    const __bf16* __restrict__ wp2, const float* __restrict__ bp2,
    float* __restrict__ out) {
  // Activations (bf16, row-major, 16-byte aligned rows):
  __shared__ __align__(16) __bf16 c0[16][352];   // [x_filled | hd0]   (layer0 comb)
  __shared__ __align__(16) __bf16 rh0[16][256];  // r0 * hd0
  __shared__ __align__(16) __bf16 c1[16][512];   // [h0 | h1]          (layer1 comb)
  __shared__ __align__(16) __bf16 rh1[16][256];  // r1 * h1

  const int tid   = threadIdx.x;
  const int lane  = tid & 31;
  const int w     = tid >> 5;
  const int bt    = blockIdx.x;
  const int nt0   = 2 * w, nt1 = nt0 + 1;
  const int nl    = lane & 15;
  const int col0  = nt0 * 16 + nl;
  const int col1  = nt1 * 16 + nl;
  const int mbase = (lane >> 4) * 8;   // C-fragment rows: mbase..mbase+7

  // h0 = h1 = 0
  for (int j = tid; j < 16 * 512; j += 256) c1[j >> 9][j & 511] = (__bf16)0.f;
  __syncthreads();

  for (int t = 0; t < 512; ++t) {
    // Opaque per-iteration zero added to every weight-tile index: keeps the
    // (loop-invariant) B-fragment loads inside the t-loop (no hoist+spill),
    // while pointers remain kernarg-derived -> global_load_b128 w/ SGPR base.
    const size_t zo = opaque_zero();

    // --- stage 1a: x_filled tile -> c0[:, 0:96) ---------------------------
    {
      const uint32_t* src = reinterpret_cast<const uint32_t*>(
          xf + (((long)t * 16 + bt) * 16) * 96);
#pragma unroll
      for (int j = tid; j < 768; j += 256) {
        const int m = j / 48, cp = j % 48;
        *reinterpret_cast<uint32_t*>(&c0[m][cp * 2]) = src[j];
      }
      if (t + 1 < 512 && tid < 192) {   // warm L2 for next step (global_prefetch_b8)
        const char* nx = reinterpret_cast<const char*>(
            xf + (((long)(t + 1) * 16 + bt) * 16) * 96);
        __builtin_prefetch(nx + tid * 16, 0, 1);
      }
    }
    // --- stage 1b: hd0 = exp(-relu(md*gh0)) * h0 -> c0[:, 96:352) ---------
    {
      const int m = tid & 15;
      const float md = md0[(bt * 16 + m) * 512 + t];
#pragma unroll
      for (int i = 0; i < 16; ++i) {
        const int h = ((tid >> 4) << 4) + i;
        const float hd = __expf(-fmaxf(md * gh0[h], 0.f)) * (float)c1[m][h];
        c0[m][96 + h] = (__bf16)hd;
      }
    }
    __syncthreads();

    // --- stage 2: layer0 r, z over K=352 ----------------------------------
    v8f ar0 = zero8(), az0 = zero8(), ar1 = zero8(), az1 = zero8();
    for (int kt = 0; kt < 11; ++kt) {
      const v16bf a = make_a(&c0[0][0], 352, kt * 32);
      ar0 = WMMA_BF16(a, load_b(wr0 + zo + (size_t)(kt * 16 + nt0) * 512), ar0);
      az0 = WMMA_BF16(a, load_b(wz0 + zo + (size_t)(kt * 16 + nt0) * 512), az0);
      ar1 = WMMA_BF16(a, load_b(wr0 + zo + (size_t)(kt * 16 + nt1) * 512), ar1);
      az1 = WMMA_BF16(a, load_b(wz0 + zo + (size_t)(kt * 16 + nt1) * 512), az1);
    }
    v8f z0 = zero8(), z1 = zero8();
    {
      const float brv0 = br0[col0], brv1 = br0[col1];
      const float bzv0 = bz0[col0], bzv1 = bz0[col1];
#pragma unroll
      for (int i = 0; i < 8; ++i) {
        const float r0 = sigmoidf_(ar0[i] + brv0);
        const float r1 = sigmoidf_(ar1[i] + brv1);
        z0[i] = sigmoidf_(az0[i] + bzv0);
        z1[i] = sigmoidf_(az1[i] + bzv1);
        rh0[mbase + i][col0] = (__bf16)(r0 * (float)c0[mbase + i][96 + col0]);
        rh0[mbase + i][col1] = (__bf16)(r1 * (float)c0[mbase + i][96 + col1]);
      }
    }
    __syncthreads();

    // --- stage 3: layer0 h_tilde, combine -> c1[:, 0:256) -----------------
    v8f ah0 = zero8(), ah1 = zero8();
    for (int kt = 0; kt < 3; ++kt) {           // x_filled part
      const v16bf a = make_a(&c0[0][0], 352, kt * 32);
      ah0 = WMMA_BF16(a, load_b(wh0 + zo + (size_t)(kt * 16 + nt0) * 512), ah0);
      ah1 = WMMA_BF16(a, load_b(wh0 + zo + (size_t)(kt * 16 + nt1) * 512), ah1);
    }
    for (int kt = 3; kt < 11; ++kt) {          // r*hd0 part
      const v16bf a = make_a(&rh0[0][0], 256, (kt - 3) * 32);
      ah0 = WMMA_BF16(a, load_b(wh0 + zo + (size_t)(kt * 16 + nt0) * 512), ah0);
      ah1 = WMMA_BF16(a, load_b(wh0 + zo + (size_t)(kt * 16 + nt1) * 512), ah1);
    }
    {
      const float bhv0 = bh0[col0], bhv1 = bh0[col1];
#pragma unroll
      for (int i = 0; i < 8; ++i) {
        const float hd0v = (float)c0[mbase + i][96 + col0];
        const float hd1v = (float)c0[mbase + i][96 + col1];
        const float th0 = tanhf(ah0[i] + bhv0);
        const float th1 = tanhf(ah1[i] + bhv1);
        c1[mbase + i][col0] = (__bf16)((1.f - z0[i]) * hd0v + z0[i] * th0);
        c1[mbase + i][col1] = (__bf16)((1.f - z1[i]) * hd1v + z1[i] * th1);
      }
    }
    __syncthreads();

    // --- stage 4: layer1 r, z over K=512 (comb = [h0_new | h1]) -----------
    ar0 = zero8(); az0 = zero8(); ar1 = zero8(); az1 = zero8();
    for (int kt = 0; kt < 16; ++kt) {
      const v16bf a = make_a(&c1[0][0], 512, kt * 32);
      ar0 = WMMA_BF16(a, load_b(wr1 + zo + (size_t)(kt * 16 + nt0) * 512), ar0);
      az0 = WMMA_BF16(a, load_b(wz1 + zo + (size_t)(kt * 16 + nt0) * 512), az0);
      ar1 = WMMA_BF16(a, load_b(wr1 + zo + (size_t)(kt * 16 + nt1) * 512), ar1);
      az1 = WMMA_BF16(a, load_b(wz1 + zo + (size_t)(kt * 16 + nt1) * 512), az1);
    }
    {
      const float brv0 = br1[col0], brv1 = br1[col1];
      const float bzv0 = bz1[col0], bzv1 = bz1[col1];
#pragma unroll
      for (int i = 0; i < 8; ++i) {
        const float r0 = sigmoidf_(ar0[i] + brv0);
        const float r1 = sigmoidf_(ar1[i] + brv1);
        z0[i] = sigmoidf_(az0[i] + bzv0);
        z1[i] = sigmoidf_(az1[i] + bzv1);
        rh1[mbase + i][col0] = (__bf16)(r0 * (float)c1[mbase + i][256 + col0]);
        rh1[mbase + i][col1] = (__bf16)(r1 * (float)c1[mbase + i][256 + col1]);
      }
    }
    __syncthreads();

    // --- stage 5: layer1 h_tilde, combine -> c1[:, 256:512) ---------------
    ah0 = zero8(); ah1 = zero8();
    for (int kt = 0; kt < 8; ++kt) {           // h0_new part
      const v16bf a = make_a(&c1[0][0], 512, kt * 32);
      ah0 = WMMA_BF16(a, load_b(wh1 + zo + (size_t)(kt * 16 + nt0) * 512), ah0);
      ah1 = WMMA_BF16(a, load_b(wh1 + zo + (size_t)(kt * 16 + nt1) * 512), ah1);
    }
    for (int kt = 8; kt < 16; ++kt) {          // r1*h1 part
      const v16bf a = make_a(&rh1[0][0], 256, (kt - 8) * 32);
      ah0 = WMMA_BF16(a, load_b(wh1 + zo + (size_t)(kt * 16 + nt0) * 512), ah0);
      ah1 = WMMA_BF16(a, load_b(wh1 + zo + (size_t)(kt * 16 + nt1) * 512), ah1);
    }
    __syncthreads();   // all reads of c1[:,256:) done before h1 overwrite
    {
      const float bhv0 = bh1[col0], bhv1 = bh1[col1];
#pragma unroll
      for (int i = 0; i < 8; ++i) {
        const float h1p0 = (float)c1[mbase + i][256 + col0];
        const float h1p1 = (float)c1[mbase + i][256 + col1];
        const float th0 = tanhf(ah0[i] + bhv0);
        const float th1 = tanhf(ah1[i] + bhv1);
        c1[mbase + i][256 + col0] = (__bf16)((1.f - z0[i]) * h1p0 + z0[i] * th0);
        c1[mbase + i][256 + col1] = (__bf16)((1.f - z1[i]) * h1p1 + z1[i] * th1);
      }
    }
    __syncthreads();
  }

  // --- head: relu(h1 @ Wp1 + bp1) @ Wp2 + bp2 -> out [B, 48] f32 ----------
  {
    v8f ap0 = zero8(), ap1 = zero8();
    for (int kt = 0; kt < 8; ++kt) {
      const v16bf a = make_a(&c1[0][0], 512, 256 + kt * 32);
      ap0 = WMMA_BF16(a, load_b(wp1 + (size_t)(kt * 16 + nt0) * 512), ap0);
      ap1 = WMMA_BF16(a, load_b(wp1 + (size_t)(kt * 16 + nt1) * 512), ap1);
    }
    const float bpv0 = bp1[col0], bpv1 = bp1[col1];
#pragma unroll
    for (int i = 0; i < 8; ++i) {
      rh0[mbase + i][col0] = (__bf16)fmaxf(ap0[i] + bpv0, 0.f);
      rh0[mbase + i][col1] = (__bf16)fmaxf(ap1[i] + bpv1, 0.f);
    }
  }
  __syncthreads();
  if (w < 3) {   // whole-wave predicate: EXEC stays all-ones inside (WMMA req.)
    v8f ao = zero8();
    for (int kt = 0; kt < 8; ++kt) {
      const v16bf a = make_a(&rh0[0][0], 256, kt * 32);
      ao = WMMA_BF16(a, load_b(wp2 + (size_t)(kt * 3 + w) * 512), ao);
    }
    const int oc = w * 16 + nl;       // 0..47
    const float bo = bp2[oc];
#pragma unroll
    for (int i = 0; i < 8; ++i)
      out[(size_t)(bt * 16 + mbase + i) * 48 + oc] = ao[i] + bo;
  }
}

// ---------------------------------------------------------------------------
extern "C" void kernel_launch(void* const* d_in, const int* in_sizes, int n_in,
                              void* d_out, int out_size, void* d_ws, size_t ws_size,
                              hipStream_t stream) {
  (void)in_sizes; (void)n_in; (void)out_size; (void)ws_size;

  const float* x     = (const float*)d_in[0];
  const float* mask  = (const float*)d_in[1];
  const float* delta = (const float*)d_in[2];
  const float* xlast = (const float*)d_in[3];
  const int*   sidx  = (const int*)d_in[4];
  const float* emb   = (const float*)d_in[5];
  const float* gx0   = (const float*)d_in[6];
  const float* gh0   = (const float*)d_in[7];
  const float* Wr0   = (const float*)d_in[8];
  const float* br0   = (const float*)d_in[9];
  const float* Wz0   = (const float*)d_in[10];
  const float* bz0   = (const float*)d_in[11];
  const float* Wh0   = (const float*)d_in[12];
  const float* bh0   = (const float*)d_in[13];
  // d_in[14]=gx1, d_in[15]=gh1 unused: layer1 sees mask=1, delta=0.
  const float* Wr1   = (const float*)d_in[16];
  const float* br1   = (const float*)d_in[17];
  const float* Wz1   = (const float*)d_in[18];
  const float* bz1   = (const float*)d_in[19];
  const float* Wh1   = (const float*)d_in[20];
  const float* bh1   = (const float*)d_in[21];
  const float* Wp1   = (const float*)d_in[22];
  const float* bp1   = (const float*)d_in[23];
  const float* Wp2   = (const float*)d_in[24];
  const float* bp2   = (const float*)d_in[25];
  float* out = (float*)d_out;

  // workspace layout (all offsets 256B-aligned)
  char* ws = (char*)d_ws;
  size_t off = 0;
  auto take = [&](size_t bytes) {
    char* p = ws + off;
    off += (bytes + 255) & ~(size_t)255;
    return p;
  };
  __bf16* xf   = (__bf16*)take(512UL * 256 * 96 * 2);   // 25.2 MB
  float*  md0  = (float*) take(256UL * 512 * 4);        // 0.5 MB
  __bf16* wr0p = (__bf16*)take(11UL * 16 * 512 * 2);
  __bf16* wz0p = (__bf16*)take(11UL * 16 * 512 * 2);
  __bf16* wh0p = (__bf16*)take(11UL * 16 * 512 * 2);
  __bf16* wr1p = (__bf16*)take(16UL * 16 * 512 * 2);
  __bf16* wz1p = (__bf16*)take(16UL * 16 * 512 * 2);
  __bf16* wh1p = (__bf16*)take(16UL * 16 * 512 * 2);
  __bf16* wp1p = (__bf16*)take(8UL * 16 * 512 * 2);
  __bf16* wp2p = (__bf16*)take(8UL * 3 * 512 * 2);

  // 1) pack weights to WMMA B-fragment layout (bf16)
  auto pack = [&](const float* W, int K, int N, __bf16* dst) {
    const int words = (K / 32) * (N / 16) * 256;
    pack_b_kernel<<<(words + 255) / 256, 256, 0, stream>>>(W, K, N, dst);
  };
  pack(Wr0, 352, 256, wr0p);
  pack(Wz0, 352, 256, wz0p);
  pack(Wh0, 352, 256, wh0p);
  pack(Wr1, 512, 256, wr1p);
  pack(Wz1, 512, 256, wz1p);
  pack(Wh1, 512, 256, wh1p);
  pack(Wp1, 256, 256, wp1p);
  pack(Wp2, 256, 48,  wp2p);

  // 2) time-parallel precompute
  {
    const long total = 512L * 256 * 96;
    xfill_kernel<<<(int)((total + 255) / 256), 256, 0, stream>>>(
        x, mask, delta, xlast, sidx, emb, gx0, xf);
    md_kernel<<<(256 * 512) / 256, 256, 0, stream>>>(delta, md0);
  }

  // 3) persistent recurrence + head (16 WGs, one 16-row batch tile each)
  grud_scan_kernel<<<16, 256, 0, stream>>>(
      xf, md0, gh0,
      wr0p, wz0p, wh0p, br0, bz0, bh0,
      wr1p, wz1p, wh1p, br1, bz1, bh1,
      wp1p, bp1, wp2p, bp2, out);
}
